// CausalSelfAttention_16587163697309
// MI455X (gfx1250) — compile-verified, matrix operand order checked by reading, not run on hardware
//
#include <hip/hip_runtime.h>

typedef __attribute__((ext_vector_type(16))) __bf16 v16bf;
typedef __attribute__((ext_vector_type(8)))  __bf16 v8bf;
typedef __attribute__((ext_vector_type(8)))  float  v8f;

union ABFrag { v16bf v; v8bf h[2]; };

constexpr int S_LEN = 2048;
constexpr int DM    = 2048;
constexpr int NH    = 16;
constexpr int HD    = 128;
constexpr int BATCH = 2;
constexpr int MTOT  = BATCH * S_LEN;   // 4096 tokens

// ---------------------------------------------------------------------------
// CDNA5 async copy global->LDS (tracked by ASYNCcnt, no VGPR data movement)
// ---------------------------------------------------------------------------
__device__ __forceinline__ void async_copy_b128(unsigned int lds_addr, const void* gaddr)
{
    asm volatile("global_load_async_to_lds_b128 %0, %1, off"
                 :: "v"(lds_addr), "v"(gaddr) : "memory");
}

__device__ __forceinline__ void wait_asynccnt0()
{
#if __has_builtin(__builtin_amdgcn_s_wait_asynccnt)
    __builtin_amdgcn_s_wait_asynccnt(0);
#else
    asm volatile("s_wait_asynccnt 0x0" ::: "memory");
#endif
}

// ---------------------------------------------------------------------------
// fp32 -> bf16 conversion
// ---------------------------------------------------------------------------
__global__ __launch_bounds__(256) void f32_to_bf16_kernel(
    const float* __restrict__ src, __bf16* __restrict__ dst, int n)
{
    int i = blockIdx.x * 256 + threadIdx.x;
    if (i < n) dst[i] = (__bf16)src[i];
}

// ---------------------------------------------------------------------------
// NT GEMM: C[m,n] = sum_k A[m,k] * W[n,k]   (A: MTOT x DM, W: DM x DM, bf16)
// Block tile 128x128 staged in LDS via async copies, double buffered.
// Block: 256 threads = 8 waves; wave tile 32x64.
// mode 0: store bf16 to [B,H,S,hd]      (Q/K buffers)
// mode 1: store bf16 to [B,H,hd,S]      (V transposed)
// mode 2: store fp32 to [MTOT, DM]      (final output)
// ---------------------------------------------------------------------------
constexpr int LDSROW = 40;              // bf16 per row: 32 data + 8 pad (80 B)
constexpr int TILE_E = 128 * LDSROW;    // elements per tile (10240 B)

__global__ __launch_bounds__(256) void proj_gemm_kernel(
    const __bf16* __restrict__ A, const __bf16* __restrict__ W,
    void* __restrict__ outp, int mode)
{
    __shared__ alignas(16) __bf16 smem[2][2][TILE_E];   // [buf][A/B][tile] = 40 KB

    const int lane = threadIdx.x & 31;
    const int wave = threadIdx.x >> 5;
    const int wm   = wave & 3;                 // 4 waves along M
    const int wn   = wave >> 2;                // 2 waves along N
    const int m0b  = blockIdx.x * 128;
    const int n0b  = blockIdx.y * 128;
    const int m0   = m0b + wm * 32;
    const int n0   = n0b + wn * 64;
    const int lr   = lane & 15;                // row within 16-tile
    const int lk8  = (lane >> 4) << 3;         // 0 or 8 (K sub-chunk select)

    const unsigned int lds_base = (unsigned int)(size_t)(&smem[0][0][0]);

    // Issue async copies for one 128x32 A-tile + 128x32 B-tile (16 KB total,
    // 1024 b128 chunks, 4 per thread, coalesced along rows).
    auto issue_tile = [&](int buf, int k0) {
#pragma unroll
        for (int i = 0; i < 4; i++) {
            int chunk = i * 256 + threadIdx.x;
            int which = chunk >> 9;            // 0 = A, 1 = B
            int c     = chunk & 511;
            int row   = c >> 2;
            int col   = (c & 3) * 8;
            const __bf16* g = (which ? W + (size_t)(n0b + row) * DM
                                     : A + (size_t)(m0b + row) * DM) + k0 + col;
            unsigned int l = lds_base +
                ((unsigned int)(buf * 2 + which) * TILE_E + row * LDSROW + col) * 2u;
            async_copy_b128(l, g);
        }
    };

    v8f acc[2][4];
#pragma unroll
    for (int i = 0; i < 2; i++)
#pragma unroll
        for (int j = 0; j < 4; j++)
#pragma unroll
            for (int e = 0; e < 8; e++) acc[i][j][e] = 0.f;

    issue_tile(0, 0);
    wait_asynccnt0();
    __syncthreads();

    int buf = 0;
    for (int k0 = 0; k0 < DM; k0 += 32) {
        if (k0 + 32 < DM) issue_tile(buf ^ 1, k0 + 32);

        const __bf16* sa = &smem[buf][0][0];
        const __bf16* sb = &smem[buf][1][0];

        ABFrag af[2], bf[4];
#pragma unroll
        for (int i = 0; i < 2; i++) {
            const __bf16* p = sa + (wm * 32 + i * 16 + lr) * LDSROW + lk8;
            af[i].h[0] = *(const v8bf*)p;
            af[i].h[1] = *(const v8bf*)(p + 16);
        }
#pragma unroll
        for (int j = 0; j < 4; j++) {
            const __bf16* p = sb + (wn * 64 + j * 16 + lr) * LDSROW + lk8;
            bf[j].h[0] = *(const v8bf*)p;
            bf[j].h[1] = *(const v8bf*)(p + 16);
        }
#pragma unroll
        for (int i = 0; i < 2; i++)
#pragma unroll
            for (int j = 0; j < 4; j++)
                acc[i][j] = __builtin_amdgcn_wmma_f32_16x16x32_bf16(
                    false, af[i].v, false, bf[j].v, (short)0, acc[i][j], false, false);

        wait_asynccnt0();
        __syncthreads();
        buf ^= 1;
    }

    // Epilogue: D layout => VGPR r holds M=r (lanes 0-15) / M=r+8 (lanes 16-31),
    // lane&15 = N column.
#pragma unroll
    for (int i = 0; i < 2; i++)
#pragma unroll
        for (int j = 0; j < 4; j++)
#pragma unroll
            for (int r = 0; r < 8; r++) {
                int m = m0 + i * 16 + r + lk8;
                int n = n0 + j * 16 + lr;
                float v = acc[i][j][r];
                if (mode == 2) {
                    ((float*)outp)[(size_t)m * DM + n] = v;
                } else {
                    int b = m >> 11, s = m & (S_LEN - 1);
                    int h = n >> 7,  d = n & (HD - 1);
                    if (mode == 0)
                        ((__bf16*)outp)[(((size_t)(b * NH + h) * S_LEN + s) * HD) + d] = (__bf16)v;
                    else
                        ((__bf16*)outp)[(((size_t)(b * NH + h) * HD + d) * S_LEN) + s] = (__bf16)v;
                }
            }
}

// ---------------------------------------------------------------------------
// RoPE (GPT-NeoX style) applied in place to [B,H,S,hd] bf16; outscale folds
// the 1/sqrt(hd) attention scale into Q.
// ---------------------------------------------------------------------------
__global__ __launch_bounds__(256) void rope_kernel(__bf16* __restrict__ X, float outscale)
{
    size_t idx = (size_t)blockIdx.x * 256 + threadIdx.x;   // B*H*S*64 threads
    int    j    = (int)(idx & 63);
    size_t rest = idx >> 6;
    int    s    = (int)(rest & (S_LEN - 1));
    size_t bh   = rest >> 11;                              // 0..B*H-1

    // inv_freq = 10000^(-j/64) = exp(-j * ln(10000)/64)
    float angle = (float)s * __expf(-(float)j * 0.14391156831212793f);
    float sn, cs;
    __sincosf(angle, &sn, &cs);

    __bf16* p = X + ((size_t)bh * S_LEN + s) * HD + j;
    float x0 = (float)p[0];
    float x1 = (float)p[64];
    p[0]  = (__bf16)((x0 * cs - x1 * sn) * outscale);
    p[64] = (__bf16)((x1 * cs + x0 * sn) * outscale);
}

// ---------------------------------------------------------------------------
// Flash attention (causal), transposed-score formulation.
// Q,K: [B,H,S,hd] bf16 (scale folded into Q). Vt: [B,H,hd,S] bf16.
// Out: [B,S,D] bf16 (input to output projection).
// Block: 256 threads = 8 waves; block covers 128 q rows; each wave owns 16.
// Per 32-wide k chunk:  S^T = K_tile x Q^T  (8 wmma),  O^T += V^T x P (8 wmma).
// ---------------------------------------------------------------------------
__global__ __launch_bounds__(256) void attn_kernel(
    const __bf16* __restrict__ Q, const __bf16* __restrict__ K,
    const __bf16* __restrict__ Vt, __bf16* __restrict__ Ob)
{
    const int lane = threadIdx.x & 31;
    const int wave = threadIdx.x >> 5;
    const int bh   = blockIdx.x >> 4;          // b*NH + h
    const int qblk = blockIdx.x & 15;
    const int q0   = qblk * 128 + wave * 16;
    const int lr   = lane & 15;
    const int lk8  = (lane >> 4) << 3;

    const __bf16* Qp = Q  + (size_t)bh * S_LEN * HD;
    const __bf16* Kp = K  + (size_t)bh * S_LEN * HD;
    const __bf16* Vp = Vt + (size_t)bh * HD * S_LEN;

    // Q fragments (B-operand): lane n holds row q0+lr along contraction c.
    ABFrag qf[4];
#pragma unroll
    for (int c = 0; c < 4; c++) {
        const __bf16* p = Qp + (size_t)(q0 + lr) * HD + c * 32 + lk8;
        qf[c].h[0] = *(const v8bf*)p;
        qf[c].h[1] = *(const v8bf*)(p + 16);
    }

    v8f o[8];                                  // O^T accumulators: 128d x 16q
#pragma unroll
    for (int t = 0; t < 8; t++)
#pragma unroll
        for (int e = 0; e < 8; e++) o[t][e] = 0.f;

    float mstat = -1e30f;
    float lstat = 0.f;
    const int qrow = q0 + lr;
    const int kend = q0 + 16;                  // causal bound (exclusive)

    for (int k0 = 0; k0 < kend; k0 += 32) {
        // S^T: two 16x16 tiles (k sub-tiles t=0,1), contraction over hd.
        float d0[8], d1[8];
#pragma unroll
        for (int t = 0; t < 2; t++) {
            v8f s;
#pragma unroll
            for (int e = 0; e < 8; e++) s[e] = 0.f;
#pragma unroll
            for (int c = 0; c < 4; c++) {
                ABFrag kf;
                const __bf16* p = Kp + (size_t)(k0 + t * 16 + lr) * HD + c * 32 + lk8;
                kf.h[0] = *(const v8bf*)p;
                kf.h[1] = *(const v8bf*)(p + 16);
                s = __builtin_amdgcn_wmma_f32_16x16x32_bf16(
                        false, kf.v, false, qf[c].v, (short)0, s, false, false);
            }
#pragma unroll
            for (int e = 0; e < 8; e++) {
                if (t == 0) d0[e] = s[e]; else d1[e] = s[e];
            }
        }

        // Causal mask: element (k,q) valid iff k <= q. k = k0 + lk8 + j (+16 for d1).
        if (k0 + 31 > q0) {
            int kb = k0 + lk8;
#pragma unroll
            for (int j = 0; j < 8; j++) {
                if (kb + j      > qrow) d0[j] = -1e30f;
                if (kb + 16 + j > qrow) d1[j] = -1e30f;
            }
        }

        // Online softmax: per-q stats; reduce regs then across half-waves.
        float mc = d0[0];
#pragma unroll
        for (int j = 1; j < 8; j++) mc = fmaxf(mc, d0[j]);
#pragma unroll
        for (int j = 0; j < 8; j++) mc = fmaxf(mc, d1[j]);
        mc = fmaxf(mc, __shfl_xor(mc, 16));

        float mn    = fmaxf(mstat, mc);
        float alpha = __expf(mstat - mn);

        float rs = 0.f;
        v16bf bp;                              // P^T packed as B-operand
#pragma unroll
        for (int j = 0; j < 8; j++) {
            float p0 = __expf(d0[j] - mn);
            float p1 = __expf(d1[j] - mn);
            rs += p0 + p1;
            bp[j]     = (__bf16)p0;            // k chunk {0..7}/{8..15}
            bp[8 + j] = (__bf16)p1;            // k chunk {16..23}/{24..31}
        }
        rs += __shfl_xor(rs, 16);
        lstat = lstat * alpha + rs;
        mstat = mn;

#pragma unroll
        for (int t = 0; t < 8; t++)
#pragma unroll
            for (int e = 0; e < 8; e++) o[t][e] *= alpha;

        // O^T += V^T x P : A = Vt rows (d along contraction k), B = packed P.
#pragma unroll
        for (int dsub = 0; dsub < 8; dsub++) {
            ABFrag vf;
            const __bf16* p = Vp + (size_t)(dsub * 16 + lr) * S_LEN + k0 + lk8;
            vf.h[0] = *(const v8bf*)p;
            vf.h[1] = *(const v8bf*)(p + 16);
            o[dsub] = __builtin_amdgcn_wmma_f32_16x16x32_bf16(
                          false, vf.v, false, bp, (short)0, o[dsub], false, false);
        }
    }

    // Normalize and store to [B,S,D] bf16. O^T tile: VGPR j -> d = dsub*16+j+lk8,
    // lane&15 -> q.
    float inv = 1.0f / lstat;
    int b = bh >> 4, h = bh & 15;
    int srow = q0 + lr;
#pragma unroll
    for (int dsub = 0; dsub < 8; dsub++)
#pragma unroll
        for (int j = 0; j < 8; j++) {
            int dd = dsub * 16 + j + lk8;
            Ob[(size_t)(b * S_LEN + srow) * DM + h * HD + dd] = (__bf16)(o[dsub][j] * inv);
        }
}

// ---------------------------------------------------------------------------
// Host-side orchestration
// ---------------------------------------------------------------------------
extern "C" void kernel_launch(void* const* d_in, const int* in_sizes, int n_in,
                              void* d_out, int out_size, void* d_ws, size_t ws_size,
                              hipStream_t stream)
{
    const float* x  = (const float*)d_in[0];
    const float* Wq = (const float*)d_in[1];
    const float* Wk = (const float*)d_in[2];
    const float* Wv = (const float*)d_in[3];
    const float* Wo = (const float*)d_in[4];

    char* ws = (char*)d_ws;
    const size_t XB_SZ = (size_t)MTOT * DM * 2;   // 16 MiB
    const size_t W_SZ  = (size_t)DM * DM * 2;     //  8 MiB
    const size_t T_SZ  = (size_t)MTOT * DM * 2;   // 16 MiB per tensor

    __bf16* xb  = (__bf16*)(ws);
    __bf16* wqb = (__bf16*)(ws + XB_SZ);
    __bf16* wkb = (__bf16*)(ws + XB_SZ + W_SZ);
    __bf16* wvb = (__bf16*)(ws + XB_SZ + 2 * W_SZ);
    __bf16* wob = (__bf16*)(ws + XB_SZ + 3 * W_SZ);
    __bf16* qb  = (__bf16*)(ws + XB_SZ + 4 * W_SZ);
    __bf16* kb  = (__bf16*)(ws + XB_SZ + 4 * W_SZ + T_SZ);
    __bf16* vtb = (__bf16*)(ws + XB_SZ + 4 * W_SZ + 2 * T_SZ);
    __bf16* atb = (__bf16*)(ws + XB_SZ + 4 * W_SZ + 3 * T_SZ);

    const int n_x = MTOT * DM;      // 8,388,608
    const int n_w = DM * DM;        // 4,194,304

    f32_to_bf16_kernel<<<n_x / 256, 256, 0, stream>>>(x,  xb,  n_x);
    f32_to_bf16_kernel<<<n_w / 256, 256, 0, stream>>>(Wq, wqb, n_w);
    f32_to_bf16_kernel<<<n_w / 256, 256, 0, stream>>>(Wk, wkb, n_w);
    f32_to_bf16_kernel<<<n_w / 256, 256, 0, stream>>>(Wv, wvb, n_w);
    f32_to_bf16_kernel<<<n_w / 256, 256, 0, stream>>>(Wo, wob, n_w);

    dim3 gemm_grid(MTOT / 128, DM / 128);       // 32 x 16
    proj_gemm_kernel<<<gemm_grid, 256, 0, stream>>>(xb, wqb, (void*)qb,  0);
    proj_gemm_kernel<<<gemm_grid, 256, 0, stream>>>(xb, wkb, (void*)kb,  0);
    proj_gemm_kernel<<<gemm_grid, 256, 0, stream>>>(xb, wvb, (void*)vtb, 1);

    const int n_rope = BATCH * NH * S_LEN * 64; // 4,194,304
    const float qscale = 0.08838834764831845f;  // 1/sqrt(128)
    rope_kernel<<<n_rope / 256, 256, 0, stream>>>(qb, qscale);
    rope_kernel<<<n_rope / 256, 256, 0, stream>>>(kb, 1.0f);

    attn_kernel<<<BATCH * NH * (S_LEN / 128), 256, 0, stream>>>(qb, kb, vtb, atb);

    proj_gemm_kernel<<<gemm_grid, 256, 0, stream>>>(atb, wob, d_out, 2);
}
